// CausalStructureGAT_80324478370387
// MI455X (gfx1250) — compile-verified, matrix operand order checked by reading, not run on hardware
//
#include <hip/hip_runtime.h>
#include <hip/hip_bf16.h>

// Problem constants (from reference)
#define BB 4
#define NN 2048
#define DD 128
#define HH 4
#define CC 64

#define LOG2E 1.4426950408889634f

typedef __attribute__((ext_vector_type(16))) __bf16 v16bf;
typedef __attribute__((ext_vector_type(8)))  float  v8f;

// 16-bit A/B-matrix element index -> K offset within a 32-wide K block:
// lane half kb = (lane<16)?0:8 ; element e: K = kb + e + (e>=8 ? 8 : 0)
__device__ __forceinline__ int kmap(int e) { return e + ((e >= 8) ? 8 : 0); }

// ---------------------------------------------------------------------------
// Kernel M: pack (causal_structure * cw != 0) into bitmask words (wave32 ballot).
// bits[i*(N/32) + j/32] bit (j%32) = keep.
// ---------------------------------------------------------------------------
__global__ void __launch_bounds__(256)
maskpack_kernel(const float* __restrict__ mask, const float* __restrict__ cw,
                unsigned* __restrict__ bits)
{
    const int idx  = blockIdx.x * blockDim.x + threadIdx.x;   // over N*N
    const float v  = mask[idx] * cw[0];
    const unsigned long long bm = __ballot(v != 0.f);         // wave32: low 32 bits
    if ((threadIdx.x & 31) == 0) bits[idx >> 5] = (unsigned)bm;
}

// ---------------------------------------------------------------------------
// Kernel F: pre-pack W [H,D,C] and gate_w [C,C] into WMMA B-fragment layout
// (bf16, 32B per lane, coalesced).  One wave per fragment.
//   Wfrag[((h*4+kblk)*4+t)*32 + lane][e] = W[h, kblk*32+kb+kmap(e), t*16+m]
//   gfrag[(kblk*4+t)*32 + lane][e]      = gate_w[t*16+m, kblk*32+kb+kmap(e)]
// ---------------------------------------------------------------------------
__global__ void __launch_bounds__(256)
packfrag_kernel(const float* __restrict__ W, const float* __restrict__ gate_w,
                __bf16* __restrict__ Wfrag, __bf16* __restrict__ gfrag)
{
    const int lane = threadIdx.x & 31;
    const int wv   = threadIdx.x >> 5;
    const int gw2  = blockIdx.x * (blockDim.x >> 5) + wv;
    const int m  = lane & 15;
    const int kb = (lane < 16) ? 0 : 8;

    if (gw2 < 64) {                         // W fragments: h,kblk,t
        const int h = gw2 >> 4, kblk = (gw2 >> 2) & 3, t = gw2 & 3;
        v16bf f;
#pragma unroll
        for (int e = 0; e < 16; ++e) {
            const int d = kblk * 32 + kb + kmap(e);
            f[e] = (__bf16)W[((size_t)h * DD + d) * CC + (t * 16 + m)];
        }
        ((v16bf*)Wfrag)[(size_t)gw2 * 32 + lane] = f;
    } else if (gw2 < 72) {                  // gate_w fragments: kblk,t
        const int q = gw2 - 64, kblk = q >> 2, t = q & 3;
        v16bf f;
#pragma unroll
        for (int e = 0; e < 16; ++e) {
            const int k = kblk * 32 + kb + kmap(e);
            f[e] = (__bf16)gate_w[(size_t)(t * 16 + m) * CC + k];
        }
        ((v16bf*)gfrag)[(size_t)q * 32 + lane] = f;
    }
}

// ---------------------------------------------------------------------------
// Kernel A: xt = x @ W (per b,h).  One wave per 32-row tile.  Emits xt as
// pre-swizzled WMMA B-fragments and s_i/s_j (scaled by log2(e) for exp2).
// ---------------------------------------------------------------------------
__global__ void __launch_bounds__(256)
proj_kernel(const float* __restrict__ x,        // [B,N,D]
            const __bf16* __restrict__ Wfrag,   // packed W fragments
            const float* __restrict__ attn,     // [H,2C]
            __bf16* __restrict__ xtS,           // swizzled xt fragments
            float*  __restrict__ s_i,           // [B*H, N]  (x log2e)
            float*  __restrict__ s_j)           // [B*H, N]  (x log2e)
{
    __shared__ float lds[8][32 * CC];           // 8KB per wave

    const int lane = threadIdx.x & 31;
    const int wv   = threadIdx.x >> 5;
    const int gw   = blockIdx.x * (blockDim.x >> 5) + wv;
    const int jb   = gw % (NN / 32);
    const int bh   = gw / (NN / 32);
    const int b    = bh / HH;
    const int h    = bh % HH;
    const int i0   = jb * 32;

    const int m  = lane & 15;
    const int kb = (lane < 16) ? 0 : 8;
    const int rbase = (lane < 16) ? 0 : 8;

    // ---- A tiles: two 16-row M-tiles, 4 K-blocks each (D = 128)
    v16bf A[2][4];
#pragma unroll
    for (int rt = 0; rt < 2; ++rt) {
        const float* xrow = x + ((size_t)b * NN + (i0 + rt * 16 + m)) * DD;
#pragma unroll
        for (int kblk = 0; kblk < 4; ++kblk)
#pragma unroll
            for (int e = 0; e < 16; ++e)
                A[rt][kblk][e] = (__bf16)xrow[kblk * 32 + kb + kmap(e)];
    }

    v8f acc[2][4];
#pragma unroll
    for (int rt = 0; rt < 2; ++rt)
#pragma unroll
        for (int t = 0; t < 4; ++t)
#pragma unroll
            for (int r = 0; r < 8; ++r) acc[rt][t][r] = 0.f;

    // ---- 32 WMMAs; W fragments are single v16bf coalesced loads
    const v16bf* wfp = (const v16bf*)Wfrag + (size_t)h * 16 * 32;
#pragma unroll
    for (int kblk = 0; kblk < 4; ++kblk) {
#pragma unroll
        for (int t = 0; t < 4; ++t) {
            const v16bf Bm = wfp[(size_t)(kblk * 4 + t) * 32 + lane];
            acc[0][t] = __builtin_amdgcn_wmma_f32_16x16x32_bf16(
                false, A[0][kblk], false, Bm, (short)0, acc[0][t], false, false);
            acc[1][t] = __builtin_amdgcn_wmma_f32_16x16x32_bf16(
                false, A[1][kblk], false, Bm, (short)0, acc[1][t], false, false);
        }
    }

    // ---- stage full 32x64 tile in LDS (f32)
#pragma unroll
    for (int rt = 0; rt < 2; ++rt)
#pragma unroll
        for (int t = 0; t < 4; ++t)
#pragma unroll
            for (int r = 0; r < 8; ++r)
                lds[wv][(rt * 16 + rbase + r) * CC + (t * 16 + m)] = acc[rt][t][r];
    __syncthreads();

    // ---- s_i / s_j: one row per lane, scaled into exp2 domain
    const float* av = attn + h * 2 * CC;
    {
        float siv = 0.f, sjv = 0.f;
#pragma unroll 8
        for (int c = 0; c < CC; ++c) {
            const float v = lds[wv][lane * CC + c];
            siv += v * av[c];
            sjv += v * av[CC + c];
        }
        s_i[(size_t)bh * NN + i0 + lane] = siv * LOG2E;
        s_j[(size_t)bh * NN + i0 + lane] = sjv * LOG2E;
    }

    // ---- emit swizzled B-fragments: lane's 16 bf16 contiguous (32B)
    v16bf* frag_out = (v16bf*)xtS + ((size_t)(bh * (NN / 32) + jb) * 4) * 32;
#pragma unroll
    for (int t = 0; t < 4; ++t) {
        v16bf frag;
#pragma unroll
        for (int e = 0; e < 16; ++e)
            frag[e] = (__bf16)lds[wv][(kb + kmap(e)) * CC + (t * 16 + m)];
        frag_out[t * 32 + lane] = frag;
    }
}

// ---------------------------------------------------------------------------
// Kernel R: exact per-row softmax max (exp2 domain):
//   rowm[bh,i] = leaky( s_i[bh,i] + max_{j: mask[i,j]!=0} s_j[bh,j] )
// (leaky_relu is monotonic, so this equals max_j score.)
// ---------------------------------------------------------------------------
__global__ void __launch_bounds__(256)
rowmax_kernel(const float* __restrict__ s_i, const float* __restrict__ s_j,
              const unsigned* __restrict__ mbits, float* __restrict__ rowm)
{
    const int idx = blockIdx.x * blockDim.x + threadIdx.x;    // over B*H*N
    const int bh  = idx >> 11;
    const int i   = idx & (NN - 1);

    const float* sjp = s_j + (size_t)bh * NN;
    const unsigned* mp = mbits + (size_t)i * (NN / 32);

    float mx = -1e30f;
    for (int k = 0; k < NN / 32; ++k) {
        unsigned w = mp[k];
        const float4* v4 = (const float4*)(sjp + k * 32);
#pragma unroll
        for (int q = 0; q < 8; ++q) {
            const float4 v = v4[q];
            mx = (w & 1u) ? fmaxf(mx, v.x) : mx; w >>= 1;
            mx = (w & 1u) ? fmaxf(mx, v.y) : mx; w >>= 1;
            mx = (w & 1u) ? fmaxf(mx, v.z) : mx; w >>= 1;
            mx = (w & 1u) ? fmaxf(mx, v.w) : mx; w >>= 1;
        }
    }
    float s = s_i[idx] + mx;
    s = fmaxf(s, 0.2f * s);                    // leaky_relu
    rowm[idx] = s;
}

// ---------------------------------------------------------------------------
// Kernel B: fused attention (fixed row max -> no rescale) + sigmoid gate +
// transposed store.  One wave per 32-row tile; two P tiles share every B frag.
// ---------------------------------------------------------------------------
__global__ void __launch_bounds__(256)
attn_kernel(const __bf16* __restrict__ xtS,     // swizzled xt fragments
            const float*  __restrict__ s_i_a,   // [B*H, N] (log2e)
            const float*  __restrict__ s_j_a,   // [B*H, N] (log2e)
            const float*  __restrict__ rowm,    // [B*H, N] fixed row max
            const unsigned* __restrict__ mbits, // [N, N/32]
            const __bf16* __restrict__ gfrag,   // packed gate_w fragments
            const float*  __restrict__ gate_b,  // [C]
            float*        __restrict__ out)     // [B, N, H*C]
{
    __shared__ float lds[8][32 * CC];           // 8KB per wave

    const int lane = threadIdx.x & 31;
    const int wv   = threadIdx.x >> 5;
    const int gw   = blockIdx.x * (blockDim.x >> 5) + wv;
    const int ib   = gw % (NN / 32);
    const int bh   = gw / (NN / 32);
    const int b    = bh / HH;
    const int h    = bh % HH;
    const int i0   = ib * 32;

    const int m  = lane & 15;
    const int kb = (lane < 16) ? 0 : 8;
    const int rbase = (lane < 16) ? 0 : 8;

    const float si0 = s_i_a[(size_t)bh * NN + i0 + m];
    const float si1 = s_i_a[(size_t)bh * NN + i0 + 16 + m];
    const float m0  = rowm[(size_t)bh * NN + i0 + m];
    const float m1  = rowm[(size_t)bh * NN + i0 + 16 + m];
    const float* sj = s_j_a + (size_t)bh * NN;
    const unsigned* mb0 = mbits + (size_t)(i0 + m) * (NN / 32);
    const unsigned* mb1 = mbits + (size_t)(i0 + 16 + m) * (NN / 32);
    const v16bf* frags = (const v16bf*)xtS + ((size_t)bh * (NN / 32) * 4) * 32;

    v8f acc[2][4];
#pragma unroll
    for (int rt = 0; rt < 2; ++rt)
#pragma unroll
        for (int t = 0; t < 4; ++t)
#pragma unroll
            for (int r = 0; r < 8; ++r) acc[rt][t][r] = 0.f;

    float lrow0 = 0.f, lrow1 = 0.f;

    for (int jblk = 0; jblk < NN / 32; ++jblk) {
        const int j0 = jblk * 32;
        if (jblk + 1 < NN / 32)
            __builtin_prefetch((const void*)(frags + (size_t)(jblk + 1) * 4 * 32 + lane), 0, 1);

        float sjv[16];
#pragma unroll
        for (int e = 0; e < 16; ++e) sjv[e] = sj[j0 + kb + kmap(e)];
        const unsigned w0 = mb0[jblk];
        const unsigned w1 = mb1[jblk];

        // ---- P tiles: exp2(score - fixed row max); no rescaling needed
        v16bf P0, P1;
        {
            float lsum = 0.f;
#pragma unroll
            for (int e = 0; e < 16; ++e) {
                float s = si0 + sjv[e];
                s = fmaxf(s, 0.2f * s);                      // leaky_relu
                const int bit = kb + kmap(e);
                s = ((w0 >> bit) & 1u) ? s : -1e30f;
                float pe = __builtin_amdgcn_exp2f(s - m0);
                pe = (s <= -1e29f) ? 0.f : pe;
                lsum += pe; P0[e] = (__bf16)pe;
            }
            lsum += __shfl_xor(lsum, 16, 32);
            lrow0 += lsum;
        }
        {
            float lsum = 0.f;
#pragma unroll
            for (int e = 0; e < 16; ++e) {
                float s = si1 + sjv[e];
                s = fmaxf(s, 0.2f * s);
                const int bit = kb + kmap(e);
                s = ((w1 >> bit) & 1u) ? s : -1e30f;
                float pe = __builtin_amdgcn_exp2f(s - m1);
                pe = (s <= -1e29f) ? 0.f : pe;
                lsum += pe; P1[e] = (__bf16)pe;
            }
            lsum += __shfl_xor(lsum, 16, 32);
            lrow1 += lsum;
        }

        // ---- 8 WMMAs: each B fragment (one v16bf load) feeds both P tiles
        const v16bf* fb = frags + (size_t)jblk * 4 * 32;
#pragma unroll
        for (int t = 0; t < 4; ++t) {
            const v16bf Bt = fb[t * 32 + lane];
            acc[0][t] = __builtin_amdgcn_wmma_f32_16x16x32_bf16(
                false, P0, false, Bt, (short)0, acc[0][t], false, false);
            acc[1][t] = __builtin_amdgcn_wmma_f32_16x16x32_bf16(
                false, P1, false, Bt, (short)0, acc[1][t], false, false);
        }
    }

    // ---- normalize by row sums, stage normalized tile in LDS
#pragma unroll
    for (int r = 0; r < 8; ++r) {
        const float l0 = __shfl(lrow0, rbase + r, 32);
        const float l1 = __shfl(lrow1, rbase + r, 32);
        const float i0v = 1.f / l0, i1v = 1.f / l1;
#pragma unroll
        for (int t = 0; t < 4; ++t) { acc[0][t][r] *= i0v; acc[1][t][r] *= i1v; }
    }
#pragma unroll
    for (int rt = 0; rt < 2; ++rt)
#pragma unroll
        for (int t = 0; t < 4; ++t)
#pragma unroll
            for (int r = 0; r < 8; ++r)
                lds[wv][(rt * 16 + rbase + r) * CC + (t * 16 + m)] = acc[rt][t][r];
    __syncthreads();

    // ---- gate = sigmoid(out @ gate_w^T + gate_b) via WMMA (16 WMMAs)
    v16bf Ag[2][2];
#pragma unroll
    for (int rt = 0; rt < 2; ++rt)
#pragma unroll
        for (int kblk = 0; kblk < 2; ++kblk)
#pragma unroll
            for (int e = 0; e < 16; ++e)
                Ag[rt][kblk][e] =
                    (__bf16)lds[wv][(rt * 16 + m) * CC + kblk * 32 + kb + kmap(e)];

    v8f g[2][4];
#pragma unroll
    for (int rt = 0; rt < 2; ++rt)
#pragma unroll
        for (int t = 0; t < 4; ++t)
#pragma unroll
            for (int r = 0; r < 8; ++r) g[rt][t][r] = 0.f;

#pragma unroll
    for (int kblk = 0; kblk < 2; ++kblk) {
#pragma unroll
        for (int t = 0; t < 4; ++t) {
            const v16bf Bg = ((const v16bf*)gfrag)[(size_t)(kblk * 4 + t) * 32 + lane];
            g[0][t] = __builtin_amdgcn_wmma_f32_16x16x32_bf16(
                false, Ag[0][kblk], false, Bg, (short)0, g[0][t], false, false);
            g[1][t] = __builtin_amdgcn_wmma_f32_16x16x32_bf16(
                false, Ag[1][kblk], false, Bg, (short)0, g[1][t], false, false);
        }
    }

    // ---- sigmoid gate * out, transposed store: out[b, i, h*C + c]
#pragma unroll
    for (int rt = 0; rt < 2; ++rt) {
#pragma unroll
        for (int t = 0; t < 4; ++t) {
            const int c = t * 16 + m;
            const float bias = gate_b[c];
#pragma unroll
            for (int r = 0; r < 8; ++r) {
                const int lrow = rt * 16 + rbase + r;
                const float ov = lds[wv][lrow * CC + c];
                const float gg = 1.f / (1.f + __expf(-(g[rt][t][r] + bias)));
                const int row = i0 + lrow;
                out[(((size_t)b * NN + row) * HH + h) * CC + c] = ov * gg;
            }
        }
    }
}

// ---------------------------------------------------------------------------
extern "C" void kernel_launch(void* const* d_in, const int* in_sizes, int n_in,
                              void* d_out, int out_size, void* d_ws, size_t ws_size,
                              hipStream_t stream) {
    const float* x    = (const float*)d_in[0];   // B*N*D
    const float* cs   = (const float*)d_in[1];   // N*N
    const float* W    = (const float*)d_in[2];   // H*D*C
    const float* attn = (const float*)d_in[3];   // H*2C
    const float* cw   = (const float*)d_in[4];   // 1
    const float* gwp  = (const float*)d_in[5];   // C*C
    const float* gb   = (const float*)d_in[6];   // C
    float* outp = (float*)d_out;

    // workspace layout (bytes):
    char* ws = (char*)d_ws;
    __bf16*   xtS   = (__bf16*)ws;                             ws += (size_t)BB * HH * CC * NN * 2;  // 4MB
    float*    s_i   = (float*)ws;                              ws += (size_t)BB * HH * NN * 4;       // 128KB
    float*    s_j   = (float*)ws;                              ws += (size_t)BB * HH * NN * 4;       // 128KB
    float*    rowm  = (float*)ws;                              ws += (size_t)BB * HH * NN * 4;       // 128KB
    unsigned* mbt   = (unsigned*)ws;                           ws += (size_t)NN * (NN / 32) * 4;     // 512KB
    __bf16*   Wfrag = (__bf16*)ws;                             ws += (size_t)HH * DD * CC * 2;       // 64KB
    __bf16*   gfrag = (__bf16*)ws;                             ws += (size_t)CC * CC * 2;            // 8KB

    maskpack_kernel<<<(NN * NN) / 256, 256, 0, stream>>>(cs, cw, mbt);
    packfrag_kernel<<<9, 256, 0, stream>>>(W, gwp, Wfrag, gfrag);

    const int waves  = BB * HH * (NN / 32);          // 1024
    const int blocks = waves / 8;                    // 128

    proj_kernel<<<blocks, 256, 0, stream>>>(x, Wfrag, attn, xtS, s_i, s_j);
    rowmax_kernel<<<(BB * HH * NN) / 256, 256, 0, stream>>>(s_i, s_j, mbt, rowm);
    attn_kernel<<<blocks, 256, 0, stream>>>(xtS, s_i, s_j, rowm, mbt, gfrag, gb, outp);
}
